// HeadAttention_3530463117519
// MI455X (gfx1250) — matrix operand order repaired and checked by reading, not compiled, not run on hardware
//
#include <hip/hip_runtime.h>
#include <math.h>

typedef _Float16 h16;
typedef __attribute__((ext_vector_type(16))) _Float16 v16h;
typedef __attribute__((ext_vector_type(8)))  _Float16 v8h;
typedef __attribute__((ext_vector_type(8)))  float    v8f;
typedef __attribute__((ext_vector_type(4)))  float    v4f;

static constexpr int BB = 8, T = 2048, C = 1024, H = 64;
static constexpr int NK = C / 32;                 // k-steps in projection

__device__ __forceinline__ v8f wmma_f16(v16h a, v16h b, v8f c) {
  return __builtin_amdgcn_wmma_f32_16x16x32_f16(false, a, false, b, (short)0, c, false, false);
}

// A-fragment from fp32 row: lane (g,m) holds row m, K = {16g..16g+15}
__device__ __forceinline__ v16h load_a_f32(const float* base, int g) {
  v4f f0 = *(const v4f*)(base + 16 * g);
  v4f f1 = *(const v4f*)(base + 16 * g + 4);
  v4f f2 = *(const v4f*)(base + 16 * g + 8);
  v4f f3 = *(const v4f*)(base + 16 * g + 12);
  v16h a;
#pragma unroll
  for (int e = 0; e < 4; ++e) {
    a[e] = (h16)f0[e]; a[4 + e] = (h16)f1[e];
    a[8 + e] = (h16)f2[e]; a[12 + e] = (h16)f3[e];
  }
  return a;
}

// ---------------------------------------------------------------------------
// Kernel 0: W[C,H] f32 -> Wt[H,C] f16 (x3) so B-fragments are contiguous.
// ---------------------------------------------------------------------------
__global__ void wt_kernel(const float* __restrict__ Wq, const float* __restrict__ Wk,
                          const float* __restrict__ Wv, h16* __restrict__ wt) {
  int idx = blockIdx.x * blockDim.x + threadIdx.x;   // [0, 3*C*H)
  int w = idx / (C * H);
  int r = idx % (C * H);
  int h = r / C, c = r % C;
  const float* W = (w == 0) ? Wq : ((w == 1) ? Wk : Wv);
  wt[(size_t)w * C * H + (size_t)h * C + c] = (h16)W[(size_t)c * H + h];
}

// Stage one 12KB W k-slab (3 mats x 64 cols x 32 k, f16) into LDS via the
// CDNA5 async-to-LDS path (tracked by ASYNCcnt). 128 threads x 6 x b128.
// chunk idx -> row = idx/4 (= w*64+col), off = idx%4 (8-half chunk of the 32 k)
// LDS byte offset = idx*16 == halfs (row*32 + off*8) -> matches the
// B-fragment read ((w*64+col)*32 + 16g).
__device__ __forceinline__ void stage_wt_async(const h16* __restrict__ wt, int kb,
                                               h16* sbuf, int tid) {
#pragma unroll
  for (int j = 0; j < 6; ++j) {
    const int idx = j * 128 + tid;
    const int row = idx >> 2, off = idx & 3;
    unsigned long long ga =
        (unsigned long long)(size_t)(wt + (size_t)row * C + kb + off * 8);
    unsigned lo = (unsigned)(size_t)sbuf + (unsigned)(idx * 16);
    asm volatile("global_load_async_to_lds_b128 %0, %1, off"
                 :: "v"(lo), "v"(ga) : "memory");
  }
}

// ---------------------------------------------------------------------------
// Kernel 1: q/k/v projections. 4 waves/block, 1 wave = one 16-row tile.
// W slabs double-buffered in LDS (async copies overlap WMMAs); A fragments
// pipelined one step ahead; all 12 B-fragments loaded into distinct registers
// before the WMMA burst so ds latency pipelines under the matrix pipe.
// ---------------------------------------------------------------------------
__global__ __launch_bounds__(128) void qkv_kernel(const float* __restrict__ x,
                                                  const h16* __restrict__ wt,
                                                  h16* __restrict__ qh,
                                                  h16* __restrict__ kh,
                                                  h16* __restrict__ vT) {
  __shared__ __align__(32) h16 sb[2][3 * 64 * 32];   // 2 x 12KB
  const int tid = threadIdx.x;
  const int wave = tid >> 5, lane = tid & 31;
  const int g = lane >> 4, m = lane & 15;
  const long r0 = ((long)blockIdx.x * 4 + wave) * 16;   // global row, [0, B*T)

  v8f acc[3][4] = {};
  const float* xrow = x + (r0 + m) * (long)C;

  stage_wt_async(wt, 0, &sb[0][0], tid);
  v16h a_cur = load_a_f32(xrow, g);

  for (int i = 0; i < NK; ++i) {
    const int kb = i * 32;
    asm volatile("s_wait_asynccnt 0" ::: "memory");  // my slab writes done
    __syncthreads();                                  // everyone's slab done
    if (i + 1 < NK) stage_wt_async(wt, kb + 32, &sb[(i + 1) & 1][0], tid);

    v16h a_next = a_cur;                              // prefetch next A frag
    if (i + 1 < NK) a_next = load_a_f32(xrow + kb + 32, g);

    const h16* sbase = &sb[i & 1][0];
    v16h barr[12];                                    // all B frags up front
#pragma unroll
    for (int f = 0; f < 12; ++f)
      barr[f] = *(const v16h*)(sbase +
                 (((f >> 2) * 64 + (f & 3) * 16 + m) * 32 + 16 * g));
#pragma unroll
    for (int w = 0; w < 3; ++w)
#pragma unroll
      for (int nt = 0; nt < 4; ++nt)
        acc[w][nt] = wmma_f16(a_cur, barr[w * 4 + nt], acc[w][nt]);
    a_cur = a_next;
  }

  const int bidx = (int)(r0 / T);
  const int t0   = (int)(r0 % T);
#pragma unroll
  for (int nt = 0; nt < 4; ++nt) {
    const int col = nt * 16 + m;
#pragma unroll
    for (int e = 0; e < 8; ++e) {                 // C layout: row = e + 8g
      long row = r0 + e + 8 * g;
      qh[row * H + col] = (h16)acc[0][nt][e];
      kh[row * H + col] = (h16)acc[1][nt][e];
    }
    v8h vv;                                        // vT: contiguous 16B store
#pragma unroll
    for (int e = 0; e < 8; ++e) vv[e] = (h16)acc[2][nt][e];
    *(v8h*)(vT + ((size_t)bidx * H + col) * T + t0 + 8 * g) = vv;
  }
}

// ---------------------------------------------------------------------------
// Kernel 2: causal flash attention. 4 waves/block, 1 wave = one 32-query tile
// (two 16-row A fragments sharing every K/V B-fragment). Key blocks of 32.
// V fragments are fetched right after the S-WMMAs so the softmax hides their
// latency.
// ---------------------------------------------------------------------------
__global__ __launch_bounds__(128) void attn_kernel(const h16* __restrict__ qh,
                                                   const h16* __restrict__ kh,
                                                   const h16* __restrict__ vT,
                                                   float* __restrict__ out) {
  __shared__ __align__(32) _Float16 pbuf[4][2 * 16 * 32];
  const int wave = threadIdx.x >> 5;
  const int lane = threadIdx.x & 31;
  const int g = lane >> 4, m = lane & 15;
  const int qt32 = blockIdx.x * 4 + wave;        // 32-query tile id
  const int b = qt32 / (T / 32);
  const int row0 = (qt32 % (T / 32)) * 32;
  _Float16* pb = pbuf[wave];

  // Q A-fragments: [qtile][h-half]
  v16h aq[2][2];
#pragma unroll
  for (int q = 0; q < 2; ++q) {
    const h16* qrow = qh + ((size_t)b * T + row0 + q * 16 + m) * H;
    v8h l0 = *(const v8h*)(qrow + 16 * g);
    v8h h0 = *(const v8h*)(qrow + 8 + 16 * g);
    v8h l1 = *(const v8h*)(qrow + 32 + 16 * g);
    v8h h1 = *(const v8h*)(qrow + 40 + 16 * g);
#pragma unroll
    for (int e = 0; e < 8; ++e) {
      aq[q][0][e] = l0[e]; aq[q][0][8 + e] = h0[e];
      aq[q][1][e] = l1[e]; aq[q][1][8 + e] = h1[e];
    }
  }

  v8f oacc[2][4] = {};
  float mrow[2][8], lrow[2][8];
#pragma unroll
  for (int q = 0; q < 2; ++q)
#pragma unroll
    for (int e = 0; e < 8; ++e) { mrow[q][e] = -INFINITY; lrow[q][e] = 0.0f; }

  const h16* kbase = kh + (size_t)b * T * H;
  const h16* vbase = vT + (size_t)b * H * T;
  const float sscale = 0.03125f;                  // C^-0.5 = 1/32
  const int nblocks = (row0 + 31) / 32 + 1;

  for (int jb = 0; jb < nblocks; ++jb) {
    const int key0 = jb * 32;
    if (jb + 1 < nblocks)                         // global_prefetch next K block
      __builtin_prefetch(kbase + (size_t)(key0 + 32 + m) * H, 0, 0);

    // K B-fragments, shared by both query tiles
    const h16* kr0 = kbase + (size_t)(key0 + m) * H;
    const h16* kr1 = kbase + (size_t)(key0 + 16 + m) * H;
    v16h b0a = *(const v16h*)(kr0 + 16 * g);
    v16h b0b = *(const v16h*)(kr0 + 32 + 16 * g);
    v16h b1a = *(const v16h*)(kr1 + 16 * g);
    v16h b1b = *(const v16h*)(kr1 + 32 + 16 * g);

    v8f s[2][2];
#pragma unroll
    for (int q = 0; q < 2; ++q) {
      v8f s0 = {}, s1 = {};
      s0 = wmma_f16(aq[q][0], b0a, s0); s0 = wmma_f16(aq[q][1], b0b, s0);
      s1 = wmma_f16(aq[q][0], b1a, s1); s1 = wmma_f16(aq[q][1], b1b, s1);
      s[q][0] = s0; s[q][1] = s1;
    }

    // issue V fragment loads now; softmax below hides their latency
    v16h vf[4];
#pragma unroll
    for (int nt = 0; nt < 4; ++nt)
      vf[nt] = *(const v16h*)(vbase + (size_t)(nt * 16 + m) * T + key0 + 16 * g);

#pragma unroll
    for (int q = 0; q < 2; ++q) {
      const int rbase = row0 + q * 16;
#pragma unroll
      for (int e = 0; e < 8; ++e) {
        const int rowg = rbase + e + 8 * g;
        float v0 = s[q][0][e] * sscale, v1 = s[q][1][e] * sscale;
        if (key0 + m > rowg)      v0 = -INFINITY;   // branchless causal mask
        if (key0 + 16 + m > rowg) v1 = -INFINITY;
        float rm = fmaxf(v0, v1);
        rm = fmaxf(rm, __shfl_xor(rm, 1, 32));
        rm = fmaxf(rm, __shfl_xor(rm, 2, 32));
        rm = fmaxf(rm, __shfl_xor(rm, 4, 32));
        rm = fmaxf(rm, __shfl_xor(rm, 8, 32));
        float mn = fmaxf(mrow[q][e], rm);
        float alpha = __expf(mrow[q][e] - mn);
        float p0 = __expf(v0 - mn);
        float p1 = __expf(v1 - mn);
        float rs = p0 + p1;
        rs += __shfl_xor(rs, 1, 32);
        rs += __shfl_xor(rs, 2, 32);
        rs += __shfl_xor(rs, 4, 32);
        rs += __shfl_xor(rs, 8, 32);
        lrow[q][e] = lrow[q][e] * alpha + rs;
        mrow[q][e] = mn;
        oacc[q][0][e] *= alpha; oacc[q][1][e] *= alpha;
        oacc[q][2][e] *= alpha; oacc[q][3][e] *= alpha;
        const int r = e + 8 * g;                  // transpose P through LDS
        pb[q * 512 + r * 32 + m]      = (h16)p0;
        pb[q * 512 + r * 32 + 16 + m] = (h16)p1;
      }
    }
    asm volatile("s_wait_dscnt 0" ::: "memory");   // per-wave LDS RAW fence

    v16h ap[2];
#pragma unroll
    for (int q = 0; q < 2; ++q) {
      v8h lo = *(const v8h*)(pb + q * 512 + m * 32 + 16 * g);
      v8h hi = *(const v8h*)(pb + q * 512 + m * 32 + 8 + 16 * g);
#pragma unroll
      for (int e = 0; e < 8; ++e) { ap[q][e] = lo[e]; ap[q][8 + e] = hi[e]; }
    }
#pragma unroll
    for (int nt = 0; nt < 4; ++nt) {
      oacc[0][nt] = wmma_f16(ap[0], vf[nt], oacc[0][nt]);
      oacc[1][nt] = wmma_f16(ap[1], vf[nt], oacc[1][nt]);
    }
  }

  // epilogue: out = O / l, fp32 [B,T,H]
#pragma unroll
  for (int q = 0; q < 2; ++q) {
    float* orow = out + ((size_t)b * T + row0 + q * 16) * H;
#pragma unroll
    for (int nt = 0; nt < 4; ++nt) {
      const int col = nt * 16 + m;
#pragma unroll
      for (int e = 0; e < 8; ++e)
        orow[(size_t)(e + 8 * g) * H + col] = oacc[q][nt][e] / lrow[q][e];
    }
  }
}

// ---------------------------------------------------------------------------
extern "C" void kernel_launch(void* const* d_in, const int* in_sizes, int n_in,
                              void* d_out, int out_size, void* d_ws, size_t ws_size,
                              hipStream_t stream) {
  const float* x  = (const float*)d_in[0];
  const float* Wq = (const float*)d_in[1];
  const float* Wk = (const float*)d_in[2];
  const float* Wv = (const float*)d_in[3];
  float* out = (float*)d_out;

  // workspace: Wt (3*H*C) | q (B*T*H) | k (B*T*H) | vT (B*H*T), all f16 (~6.7MB)
  h16* wt = (h16*)d_ws;
  h16* qh = wt + (size_t)3 * H * C;
  h16* kh = qh + (size_t)BB * T * H;
  h16* vT = kh + (size_t)BB * T * H;

  wt_kernel<<<(3 * C * H) / 256, 256, 0, stream>>>(Wq, Wk, Wv, wt);
  qkv_kernel<<<(BB * T / 16) / 4, 128, 0, stream>>>(x, wt, qh, kh, vT);
  attn_kernel<<<(BB * (T / 32)) / 4, 128, 0, stream>>>(qh, kh, vT, out);
}
